// IndividualAttention_1726576855777
// MI455X (gfx1250) — matrix operand order repaired and checked by reading, not compiled, not run on hardware
//
#include <hip/hip_runtime.h>
#include <hip/hip_bf16.h>
#include <stdint.h>
#include <stddef.h>

// ---------------------------------------------------------------------------
// Multi-head self-attention, bf16 WMMA + TDM async tiles (gfx1250 / CDNA5)
// B=32, S=1024, C=512, H=8, dh=64
// ---------------------------------------------------------------------------

typedef __bf16 v16bf __attribute__((ext_vector_type(16)));
typedef __bf16 v8bf  __attribute__((ext_vector_type(8)));
typedef float  v8f   __attribute__((ext_vector_type(8)));
typedef float  v4f   __attribute__((ext_vector_type(4)));
typedef unsigned int v4u __attribute__((ext_vector_type(4)));
typedef int    v8i   __attribute__((ext_vector_type(8)));
typedef int    v4i   __attribute__((ext_vector_type(4)));

#define BATCH 32
#define SEQ   1024
#define CH    512
#define HEADS 8
#define DH    64

static __device__ __forceinline__ v8bf ld8(const __bf16* p) {
  return *reinterpret_cast<const v8bf*>(p);
}
static __device__ __forceinline__ v16bf cat16(v8bf lo, v8bf hi) {
  return __builtin_shufflevector(lo, hi, 0, 1, 2, 3, 4, 5, 6, 7,
                                         8, 9, 10, 11, 12, 13, 14, 15);
}
static __device__ __forceinline__ v8f wmma_bf16(v16bf a, v16bf b, v8f c) {
  // (neg_a, A, neg_b, B, c_mod, C, reuse_a, reuse_b)
  return __builtin_amdgcn_wmma_f32_16x16x32_bf16(false, a, false, b,
                                                 (short)0, c, false, false);
}
static __device__ __forceinline__ float rowmax16(float v) {
  v = fmaxf(v, __shfl_xor(v, 8, 32));
  v = fmaxf(v, __shfl_xor(v, 4, 32));
  v = fmaxf(v, __shfl_xor(v, 2, 32));
  v = fmaxf(v, __shfl_xor(v, 1, 32));
  return v;
}
static __device__ __forceinline__ float rowsum16(float v) {
  v += __shfl_xor(v, 8, 32);
  v += __shfl_xor(v, 4, 32);
  v += __shfl_xor(v, 2, 32);
  v += __shfl_xor(v, 1, 32);
  return v;
}

// ---------------------------------------------------------------------------
// Tensor Data Mover: 2-D bf16 tile (tile1 rows x tile0 cols) -> LDS, row-major.
// D# per CDNA5 ISA ch.8: group0 {count, lds_addr, global_addr, type=2},
// group1 {data_size=2B, tensor_dim0/1, tile_dim0/1, tensor_dim0_stride}.
// ---------------------------------------------------------------------------
static __device__ __forceinline__ void
tdm_load_2d_bf16(const __bf16* lds_dst, const __bf16* gsrc,
                 uint32_t dim0, uint32_t dim1, uint32_t tile0, uint32_t tile1,
                 uint32_t stride0) {
  const uint32_t lds_off = (uint32_t)(uintptr_t)lds_dst;  // addr[31:0] = LDS offset
  const uint64_t ga      = (uint64_t)(uintptr_t)gsrc;
  v4u g0;
  g0[0] = 1u;                                             // count=1, user mode
  g0[1] = lds_off;                                        // lds_addr [63:32]
  g0[2] = (uint32_t)ga;                                   // global_addr lo
  g0[3] = (uint32_t)((ga >> 32) & 0x1FFFFFFu) | (2u << 30);  // ga hi | type=2
  v8i g1;
  g1[0] = (int)(1u << 16);                                // data_size=1 (2 bytes)
  g1[1] = (int)((dim0 & 0xFFFFu) << 16);                  // tensor_dim0[15:0]
  g1[2] = (int)(((dim0 >> 16) & 0xFFFFu) | ((dim1 & 0xFFFFu) << 16));
  g1[3] = (int)(((dim1 >> 16) & 0xFFFFu) | ((tile0 & 0xFFFFu) << 16));
  g1[4] = (int)(tile1 & 0xFFFFu);                         // tile_dim1, tile_dim2=0
  g1[5] = (int)stride0;                                   // tensor_dim0_stride lo
  g1[6] = 0;                                              // stride hi / dim1_stride
  g1[7] = 0;
  v4i gz = {0, 0, 0, 0};
#if defined(__clang_major__) && (__clang_major__ >= 23)
  v8i gz8 = {0, 0, 0, 0, 0, 0, 0, 0};
  __builtin_amdgcn_tensor_load_to_lds(g0, g1, gz, gz, gz8, 0);
#else
  __builtin_amdgcn_tensor_load_to_lds(g0, g1, gz, gz, 0);
#endif
}

// ---------------------------------------------------------------------------
// Kernel 0: fp32 -> bf16 conversion (8 elements / thread, b128 traffic)
// ---------------------------------------------------------------------------
__global__ __launch_bounds__(256)
void k_cvt_bf16(const float* __restrict__ in, __bf16* __restrict__ out, int n8) {
  int i = blockIdx.x * blockDim.x + threadIdx.x;
  if (i >= n8) return;
  const v4f* p = reinterpret_cast<const v4f*>(in) + 2 * (size_t)i;
  v4f a = p[0];
  v4f b = p[1];
  v8bf o;
#pragma unroll
  for (int j = 0; j < 4; ++j) {
    o[j]     = (__bf16)a[j];
    o[4 + j] = (__bf16)b[j];
  }
  reinterpret_cast<v8bf*>(out)[i] = o;
}

// ---------------------------------------------------------------------------
// Kernel 1: fused QKV projection.  Weight tiles (64n x 32k, shared by all 8
// waves of the block) are TDM-streamed into double-buffered LDS; X A-fragments
// are per-wave and software-pipelined from global.
// ---------------------------------------------------------------------------
__global__ __launch_bounds__(256)
void k_qkv_gemm(const __bf16* __restrict__ X,
                const __bf16* __restrict__ Wq, const __bf16* __restrict__ Wk,
                const __bf16* __restrict__ Wv,
                __bf16* __restrict__ Q, __bf16* __restrict__ K,
                __bf16* __restrict__ V) {
  __shared__ __bf16 Wt[2][3][64 * 32];  // [buf][q/k/v][nloc*32 + kloc]  24 KB

  const int lane = threadIdx.x & 31;
  const int wave = threadIdx.x >> 5;
  const int mrow = lane & 15;
  const int g    = lane >> 4;
  const int M0   = blockIdx.x * 128 + wave * 16;
  const int N0   = blockIdx.y * 64;

  v8f accQ[4] = {}, accK[4] = {}, accV[4] = {};
  const __bf16* xrow = X + (size_t)(M0 + mrow) * CH;

  if (wave == 0) {  // DMA first weight k-slab
    tdm_load_2d_bf16(&Wt[0][0][0], Wq + (size_t)N0 * CH, CH, CH, 32, 64, CH);
    tdm_load_2d_bf16(&Wt[0][1][0], Wk + (size_t)N0 * CH, CH, CH, 32, 64, CH);
    tdm_load_2d_bf16(&Wt[0][2][0], Wv + (size_t)N0 * CH, CH, CH, 32, 64, CH);
  }
  v16bf a = cat16(ld8(xrow + 16 * g), ld8(xrow + 8 + 16 * g));
  for (int kk = 0; kk < CH; kk += 32) {
    const int buf = (kk >> 5) & 1;
    __syncthreads();  // readers of buf^1 (previous slab) are done
    if (wave == 0) {
      if (kk + 32 < CH) {
        const size_t wo = (size_t)N0 * CH + kk + 32;
        tdm_load_2d_bf16(&Wt[buf ^ 1][0][0], Wq + wo, CH, CH, 32, 64, CH);
        tdm_load_2d_bf16(&Wt[buf ^ 1][1][0], Wk + wo, CH, CH, 32, 64, CH);
        tdm_load_2d_bf16(&Wt[buf ^ 1][2][0], Wv + wo, CH, CH, 32, 64, CH);
        __builtin_amdgcn_s_wait_tensorcnt(3);  // current slab's 3 DMAs landed
      } else {
        __builtin_amdgcn_s_wait_tensorcnt(0);
      }
    }
    __syncthreads();  // publish slab kk to all waves

    const v16bf acur = a;
    if (kk + 32 < CH)  // pipeline next A-fragment behind this step's WMMAs
      a = cat16(ld8(xrow + kk + 32 + 16 * g), ld8(xrow + kk + 40 + 16 * g));
#pragma unroll
    for (int t = 0; t < 4; ++t) {
      const int nl = t * 16 + mrow;  // local output column
      const __bf16* wq = &Wt[buf][0][nl * 32 + 16 * g];
      const __bf16* wk = &Wt[buf][1][nl * 32 + 16 * g];
      const __bf16* wv = &Wt[buf][2][nl * 32 + 16 * g];
      v16bf bq = cat16(ld8(wq), ld8(wq + 8));
      v16bf bk = cat16(ld8(wk), ld8(wk + 8));
      v16bf bv = cat16(ld8(wv), ld8(wv + 8));
      accQ[t] = wmma_bf16(acur, bq, accQ[t]);
      accK[t] = wmma_bf16(acur, bk, accK[t]);
      accV[t] = wmma_bf16(acur, bv, accV[t]);
    }
  }
#pragma unroll
  for (int t = 0; t < 4; ++t) {
    const int col = N0 + t * 16 + mrow;
#pragma unroll
    for (int r = 0; r < 8; ++r) {
      const size_t row = (size_t)(M0 + r + 8 * g);
      Q[row * CH + col] = (__bf16)accQ[t][r];
      K[row * CH + col] = (__bf16)accK[t][r];
      V[row * CH + col] = (__bf16)accV[t][r];
    }
  }
}

// ---------------------------------------------------------------------------
// Kernel 2: flash attention with TDM double-buffered K/V tiles in LDS.
// ---------------------------------------------------------------------------
__global__ __launch_bounds__(256)
void k_flash_attn(const __bf16* __restrict__ Q, const __bf16* __restrict__ K,
                  const __bf16* __restrict__ V, __bf16* __restrict__ O) {
  __shared__ __bf16 Kc[2][64 * 64];     // [key][d] chunks, double-buffered 16 KB
  __shared__ __bf16 Vraw[2][64 * 64];   // [key][d] chunks, double-buffered 16 KB
  __shared__ __bf16 Vt[64 * 64];        // [d][key] transposed current chunk 8 KB
  __shared__ __bf16 P[8][16 * 64];      // per-wave probabilities          16 KB

  const int b    = blockIdx.z;
  const int h    = blockIdx.y;
  const int lane = threadIdx.x & 31;
  const int wave = threadIdx.x >> 5;
  const int mrow = lane & 15;
  const int g    = lane >> 4;
  const int s0   = blockIdx.x * 128 + wave * 16;
  const size_t base = ((size_t)b * SEQ) * CH + (size_t)h * DH;
  const float scale = 0.125f;  // dh^-0.5

  v16bf aq0, aq1;
  {
    const __bf16* q = Q + base + (size_t)(s0 + mrow) * CH;
    aq0 = cat16(ld8(q + 16 * g), ld8(q + 8 + 16 * g));
    aq1 = cat16(ld8(q + 32 + 16 * g), ld8(q + 40 + 16 * g));
  }

  v8f   oacc[4] = {};
  float mi[8], li[8];
#pragma unroll
  for (int r = 0; r < 8; ++r) { mi[r] = -1e30f; li[r] = 0.0f; }

  if (wave == 0) {  // DMA chunk 0
    tdm_load_2d_bf16(&Kc[0][0],   K + base, CH, BATCH * SEQ, 64, 64, CH);
    tdm_load_2d_bf16(&Vraw[0][0], V + base, CH, BATCH * SEQ, 64, 64, CH);
  }

  for (int c = 0; c < 16; ++c) {
    const int sk0 = c * 64;
    const int buf = c & 1;
    __syncthreads();  // all readers of buffer buf^1 (iteration c-1) are done
    if (wave == 0) {
      if (c + 1 < 16) {
        const size_t nb = base + (size_t)(sk0 + 64) * CH;
        tdm_load_2d_bf16(&Kc[buf ^ 1][0],   K + nb, CH, BATCH * SEQ, 64, 64, CH);
        tdm_load_2d_bf16(&Vraw[buf ^ 1][0], V + nb, CH, BATCH * SEQ, 64, 64, CH);
        __builtin_amdgcn_s_wait_tensorcnt(2);  // chunk c's 2 DMAs have landed
      } else {
        __builtin_amdgcn_s_wait_tensorcnt(0);
      }
    }
    if (c + 2 < 16)  // warm GL2 ahead of the next-next TDM (global_prefetch_b8)
      __builtin_prefetch(K + base + (size_t)(sk0 + 128 + (threadIdx.x >> 2)) * CH, 0, 1);
    __syncthreads();  // publish chunk c LDS data to all waves

    // transpose Vraw[buf] -> Vt   (LDS -> regs -> LDS)
    {
      const int key = threadIdx.x >> 2;
      const int d0  = (threadIdx.x & 3) * 16;
      const __bf16* vp = &Vraw[buf][key * 64 + d0];
      v8bf lo = ld8(vp);
      v8bf hi = ld8(vp + 8);
#pragma unroll
      for (int i = 0; i < 8; ++i) {
        Vt[(d0 + i) * 64 + key]     = lo[i];
        Vt[(d0 + 8 + i) * 64 + key] = hi[i];
      }
    }
    __syncthreads();  // Vt ready

    // ---- scores S = Q K^T, K B-fragments straight from LDS ----
    v8f s[4] = {};
#pragma unroll
    for (int t = 0; t < 4; ++t) {
      const __bf16* kp = &Kc[buf][(t * 16 + mrow) * 64];
      v16bf b0 = cat16(ld8(kp + 16 * g), ld8(kp + 16 * g + 8));
      s[t] = wmma_bf16(aq0, b0, s[t]);
      v16bf b1 = cat16(ld8(kp + 32 + 16 * g), ld8(kp + 32 + 16 * g + 8));
      s[t] = wmma_bf16(aq1, b1, s[t]);
    }

    // ---- online softmax ----
    float alpha[8];
#pragma unroll
    for (int r = 0; r < 8; ++r) {
      float mx = fmaxf(fmaxf(s[0][r], s[1][r]), fmaxf(s[2][r], s[3][r])) * scale;
      mx = rowmax16(mx);
      const float mnew = fmaxf(mi[r], mx);
      alpha[r] = __expf(mi[r] - mnew);
      mi[r]    = mnew;
    }
#pragma unroll
    for (int r = 0; r < 8; ++r) {
      float rs = 0.0f;
#pragma unroll
      for (int t = 0; t < 4; ++t) {
        const float p = __expf(s[t][r] * scale - mi[r]);
        rs += p;
        P[wave][(r + 8 * g) * 64 + t * 16 + mrow] = (__bf16)p;
      }
      rs = rowsum16(rs);
      li[r] = li[r] * alpha[r] + rs;
#pragma unroll
      for (int dt = 0; dt < 4; ++dt) oacc[dt][r] *= alpha[r];
    }
    asm volatile("s_wait_dscnt 0" ::: "memory");  // P stores -> P loads (same wave)

    // ---- O += P V  (A = P from LDS, B = V^T from LDS) ----
    const __bf16* pw = &P[wave][0];
    v16bf ap0 = cat16(ld8(pw + mrow * 64 + 16 * g),
                      ld8(pw + mrow * 64 + 8 + 16 * g));
    v16bf ap1 = cat16(ld8(pw + mrow * 64 + 32 + 16 * g),
                      ld8(pw + mrow * 64 + 40 + 16 * g));
#pragma unroll
    for (int dt = 0; dt < 4; ++dt) {
      const __bf16* vtp = &Vt[(dt * 16 + mrow) * 64];
      v16bf bv0 = cat16(ld8(vtp + 16 * g), ld8(vtp + 16 * g + 8));
      oacc[dt] = wmma_bf16(ap0, bv0, oacc[dt]);
      v16bf bv1 = cat16(ld8(vtp + 32 + 16 * g), ld8(vtp + 32 + 16 * g + 8));
      oacc[dt] = wmma_bf16(ap1, bv1, oacc[dt]);
    }
  }

#pragma unroll
  for (int r = 0; r < 8; ++r) li[r] = 1.0f / li[r];
#pragma unroll
  for (int dt = 0; dt < 4; ++dt) {
#pragma unroll
    for (int r = 0; r < 8; ++r) {
      O[base + (size_t)(s0 + r + 8 * g) * CH + dt * 16 + mrow] =
          (__bf16)(oacc[dt][r] * li[r]);
    }
  }
}

// ---------------------------------------------------------------------------
// Kernel 3: output projection  out = O @ Wo^T + bo   (fp32 out)
// Wo tiles TDM-streamed into double-buffered LDS like kernel 1.
// ---------------------------------------------------------------------------
__global__ __launch_bounds__(256)
void k_out_proj(const __bf16* __restrict__ Ob, const __bf16* __restrict__ Wo,
                const float* __restrict__ bo, float* __restrict__ out) {
  __shared__ __bf16 Wt[2][64 * 32];  // double-buffered weight slab  8 KB

  const int lane = threadIdx.x & 31;
  const int wave = threadIdx.x >> 5;
  const int mrow = lane & 15;
  const int g    = lane >> 4;
  const int M0   = blockIdx.x * 128 + wave * 16;
  const int N0   = blockIdx.y * 64;

  v8f acc[4] = {};
  const __bf16* xrow = Ob + (size_t)(M0 + mrow) * CH;

  if (wave == 0)
    tdm_load_2d_bf16(&Wt[0][0], Wo + (size_t)N0 * CH, CH, CH, 32, 64, CH);

  v16bf a = cat16(ld8(xrow + 16 * g), ld8(xrow + 8 + 16 * g));
  for (int kk = 0; kk < CH; kk += 32) {
    const int buf = (kk >> 5) & 1;
    __syncthreads();
    if (wave == 0) {
      if (kk + 32 < CH) {
        tdm_load_2d_bf16(&Wt[buf ^ 1][0], Wo + (size_t)N0 * CH + kk + 32,
                         CH, CH, 32, 64, CH);
        __builtin_amdgcn_s_wait_tensorcnt(1);
      } else {
        __builtin_amdgcn_s_wait_tensorcnt(0);
      }
    }
    __syncthreads();

    const v16bf acur = a;
    if (kk + 32 < CH)
      a = cat16(ld8(xrow + kk + 32 + 16 * g), ld8(xrow + kk + 40 + 16 * g));
#pragma unroll
    for (int t = 0; t < 4; ++t) {
      const __bf16* w = &Wt[buf][(t * 16 + mrow) * 32 + 16 * g];
      v16bf bw = cat16(ld8(w), ld8(w + 8));
      acc[t] = wmma_bf16(acur, bw, acc[t]);
    }
  }
#pragma unroll
  for (int t = 0; t < 4; ++t) {
    const int col  = N0 + t * 16 + mrow;
    const float bb = bo[col];
#pragma unroll
    for (int r = 0; r < 8; ++r) {
      const size_t row = (size_t)(M0 + r + 8 * g);
      out[row * CH + col] = acc[t][r] + bb;
    }
  }
}

// ---------------------------------------------------------------------------
// Launch
// ---------------------------------------------------------------------------
extern "C" void kernel_launch(void* const* d_in, const int* in_sizes, int n_in,
                              void* d_out, int out_size, void* d_ws,
                              size_t ws_size, hipStream_t stream) {
  const float* hs = (const float*)d_in[0];
  const float* Wq = (const float*)d_in[1];
  const float* Wk = (const float*)d_in[2];
  const float* Wv = (const float*)d_in[3];
  const float* Wo = (const float*)d_in[4];
  const float* bo = (const float*)d_in[5];
  (void)in_sizes; (void)n_in; (void)out_size; (void)ws_size;

  char* ws = (char*)d_ws;
  const size_t XEL = (size_t)BATCH * SEQ * CH;  // 16,777,216
  const size_t WEL = (size_t)CH * CH;           // 262,144
  __bf16* Xb  = (__bf16*)(ws);
  __bf16* Qb  = (__bf16*)(ws + 2 * XEL);
  __bf16* Kb  = (__bf16*)(ws + 4 * XEL);
  __bf16* Vb  = (__bf16*)(ws + 6 * XEL);
  __bf16* Ob  = (__bf16*)(ws + 8 * XEL);
  __bf16* Wqb = (__bf16*)(ws + 10 * XEL);
  __bf16* Wkb = (__bf16*)(ws + 10 * XEL + 2 * WEL);
  __bf16* Wvb = (__bf16*)(ws + 10 * XEL + 4 * WEL);
  __bf16* Wob = (__bf16*)(ws + 10 * XEL + 6 * WEL);

  {
    const int nX8 = (int)(XEL / 8);
    k_cvt_bf16<<<(nX8 + 255) / 256, 256, 0, stream>>>(hs, Xb, nX8);
    const int nW8 = (int)(WEL / 8);
    k_cvt_bf16<<<(nW8 + 255) / 256, 256, 0, stream>>>(Wq, Wqb, nW8);
    k_cvt_bf16<<<(nW8 + 255) / 256, 256, 0, stream>>>(Wk, Wkb, nW8);
    k_cvt_bf16<<<(nW8 + 255) / 256, 256, 0, stream>>>(Wv, Wvb, nW8);
    k_cvt_bf16<<<(nW8 + 255) / 256, 256, 0, stream>>>(Wo, Wob, nW8);
  }
  k_qkv_gemm<<<dim3(256, 8), 256, 0, stream>>>(Xb, Wqb, Wkb, Wvb, Qb, Kb, Vb);
  k_flash_attn<<<dim3(8, HEADS, BATCH), 256, 0, stream>>>(Qb, Kb, Vb, Ob);
  k_out_proj<<<dim3(256, 8), 256, 0, stream>>>(Ob, Wob, bo, (float*)d_out);
}
